// SerialIntervalGraphV2_32830730010774
// MI455X (gfx1250) — matrix operand order repaired and checked by reading, not compiled, not run on hardware
//
#include <hip/hip_runtime.h>
#include <hip/hip_bf16.h>

// ---------------------------------------------------------------------------
// Problem constants (from reference)
// ---------------------------------------------------------------------------
#define BB   32
#define TT   64
#define NN   2048
#define HID  256
#define HEADS 8
#define DH   32
#define BOT  64
#define ROWS (BB * NN)           // 65536 rows of the [B*N, HID] activations

// ---------------------------------------------------------------------------
// Types / helpers
// ---------------------------------------------------------------------------
typedef __attribute__((ext_vector_type(16))) __bf16    v16bf;
typedef __attribute__((ext_vector_type(8)))  float     v8f;
typedef __attribute__((ext_vector_type(4)))  unsigned  u32x4;
typedef __attribute__((ext_vector_type(8)))  int       i32x8;
typedef __attribute__((ext_vector_type(4)))  int       i32x4;

struct DualU4 { u32x4 lo, hi; };   // 32 bytes == v16bf

static __device__ __forceinline__ __bf16 f2bf(float f) {
    unsigned u = __builtin_bit_cast(unsigned, f);
    unsigned r = (u + 0x7FFFu + ((u >> 16) & 1u)) >> 16;   // round-nearest-even
    return __builtin_bit_cast(__bf16, (unsigned short)r);
}
static __device__ __forceinline__ float bf2f(__bf16 b) {
    unsigned u = ((unsigned)__builtin_bit_cast(unsigned short, b)) << 16;
    return __builtin_bit_cast(float, u);
}
static __device__ __forceinline__ float eluplus(float x) {
    return x > 0.f ? x + 1.f : __expf(x);   // elu(x)+1
}

// ---------------------------------------------------------------------------
// Tensor Data Mover: 2D tile load (R rows x 16 dwords) global -> LDS.
// Descriptor packing per CDNA5 ISA.
// ---------------------------------------------------------------------------
static __device__ __forceinline__ void tdm_load_2d(
    const void* gaddr, unsigned lds_off, unsigned tile_rows,
    unsigned row_stride_dw, bool pad)
{
    unsigned long long ga = (unsigned long long)gaddr;
    u32x4 g0;
    g0[0] = 1u;                                     // count=1, user mode
    g0[1] = lds_off;                                // lds_addr (bytes)
    g0[2] = (unsigned)(ga & 0xffffffffu);           // global_addr[31:0]
    g0[3] = (unsigned)((ga >> 32) & 0x01ffffffu)    // global_addr[56:32]
          | (2u << 30);                             // type = 2 ("image")
    i32x8 g1;
    unsigned w0 = (2u << 16);                       // data_size = 4 bytes
    if (pad) w0 |= (1u << 20)                       // pad_enable
              |  (3u << 22)                         // pad_interval: 16 dwords
              |  (3u << 25);                        // pad_amount: 4 dwords
    const unsigned td0 = 1u << 20, td1 = 1u << 20;  // tensor dims: huge
    g1[0] = (int)w0;
    g1[1] = (int)((td0 & 0xffffu) << 16);                       // tensor_dim0 lo
    g1[2] = (int)(((td0 >> 16) & 0xffffu) | ((td1 & 0xffffu) << 16));
    g1[3] = (int)(((td1 >> 16) & 0xffffu) | (16u << 16));       // tile_dim0 = 16 dw
    g1[4] = (int)(tile_rows & 0xffffu);                         // tile_dim1
    g1[5] = (int)row_stride_dw;                                 // dim0_stride lo32
    g1[6] = 0;
    g1[7] = 0;
    i32x4 z4 = {0, 0, 0, 0};
#if __clang_major__ >= 23
    i32x8 z8 = {0, 0, 0, 0, 0, 0, 0, 0};
    __builtin_amdgcn_tensor_load_to_lds(g0, g1, z4, z4, z8, 0);
#else
    __builtin_amdgcn_tensor_load_to_lds(g0, g1, z4, z4, 0);
#endif
}

// ---------------------------------------------------------------------------
// K0: weights -> bf16 *transposed* (B^T layout, row = output col, len = K),
//     adj -> bf16 (row-major, it is the A operand), softmax/sigmoid scalars.
// ---------------------------------------------------------------------------
#define S1 (HID * 3 * BOT)        // 256*192  w_qkv_low
#define S2 (3 * BOT * 3 * HID)    // 192*768  w_qkv_high
#define S3 (HID * BOT)            // 256*64   w_out_low
#define S4 (BOT * HID)            // 64*256   w_out_high
#define S5 (NN * NN)              // adj
#define STOT (S1 + S2 + S3 + S4 + S5)

__global__ __launch_bounds__(256) void prep_kernel(
    const float* __restrict__ dl, const float* __restrict__ geo,
    const float* __restrict__ wql, const float* __restrict__ wqh,
    const float* __restrict__ wol, const float* __restrict__ woh,
    const float* __restrict__ adj,
    float* __restrict__ scal,
    __bf16* __restrict__ wqlT, __bf16* __restrict__ wqhT,
    __bf16* __restrict__ wolT, __bf16* __restrict__ wohT,
    __bf16* __restrict__ adjb)
{
    int i = blockIdx.x * 256 + threadIdx.x;
    if (i == 0) {
        float mx = -1e30f;
        for (int t = 0; t <= 7; ++t) mx = fmaxf(mx, dl[t]);
        float e[8], s = 0.f;
        for (int t = 0; t <= 7; ++t) { e[t] = __expf(dl[t] - mx); s += e[t]; }
        for (int t = 0; t <= 7; ++t) scal[t] = e[t] / s;
        scal[8] = 1.f / (1.f + __expf(-geo[0]));
    }
    if (i < S1) {                                   // [256,192] -> T[192,256]
        int r = i / 192, c = i % 192;
        wqlT[c * HID + r] = f2bf(wql[i]);
    } else if (i < S1 + S2) {                       // [192,768] -> T[768,192]
        int j = i - S1, r = j / 768, c = j % 768;
        wqhT[c * (3 * BOT) + r] = f2bf(wqh[j]);
    } else if (i < S1 + S2 + S3) {                  // [256,64] -> T[64,256]
        int j = i - (S1 + S2), r = j / 64, c = j % 64;
        wolT[c * HID + r] = f2bf(wol[j]);
    } else if (i < S1 + S2 + S3 + S4) {             // [64,256] -> T[256,64]
        int j = i - (S1 + S2 + S3), r = j / 256, c = j % 256;
        wohT[c * BOT + r] = f2bf(woh[j]);
    } else if (i < STOT) {
        int j = i - (S1 + S2 + S3 + S4);
        adjb[j] = f2bf(adj[j]);
    }
}

// ---------------------------------------------------------------------------
// K1: combined = features + 0.1*(delayed * w_dp + b_dp)  -> bf16 [ROWS, 256]
// ---------------------------------------------------------------------------
__global__ __launch_bounds__(256) void combined_kernel(
    const float* __restrict__ x, const float* __restrict__ features,
    const float* __restrict__ w_dp, const float* __restrict__ b_dp,
    const float* __restrict__ scal, __bf16* __restrict__ comb)
{
    long row = blockIdx.x;              // b*N + n
    int  b   = (int)(row / NN);
    int  n   = (int)(row % NN);
    int  h   = threadIdx.x;
    float delayed = 0.f;
#pragma unroll
    for (int tau = 0; tau < 8; ++tau)
        delayed += scal[tau] * x[(long)b * TT * NN + (long)(TT - 1 - tau) * NN + n];
    float c = features[row * HID + h] + 0.1f * (delayed * w_dp[h] + b_dp[h]);
    comb[row * HID + h] = f2bf(c);
}

// ---------------------------------------------------------------------------
// Generic bf16 WMMA GEMM: C[M,Nc] = A[M,K] * B^T[Nc,K]^T (+ epilogue).
// 256 threads/block, 128 x TN output tile (TN = 64 or 128), K-step 32,
// double-buffered TDM staging (wave0 drives A DMA, wave1 drives B DMA).
// Each wave owns a 32 x (TN/2) register tile -> 2 x (TN/32) WMMAs per K-step
// (TN=128: 8 x v_wmma_f32_16x16x32_bf16 per wave per K-step).
// LDS: A rows at 64B stride; B rows TDM-padded to 80B stride (40 halves)
// for conflict-free ds_load_b128 fragment reads.
// Fragment mappings follow CDNA5 ISA §7.12.2:
//   A lane<16: K 0..7,16..23  lane>=16: K 8..15,24..31 (row = lane&15)
//   B lane<16: K 0..7,16..23  lane>=16: K 8..15,24..31 (col = lane&15)
//   C lane<16: M=j            lane>=16: M=8+j          (col = lane&15)
// Epilogues: 0 bias->bf16 | 1 QKV split | 2 geo blend->bf16 | 3 residual->f32
// ---------------------------------------------------------------------------
template<int EPI, int TN>
__global__ __launch_bounds__(256) void gemm_bf16_kernel(
    const __bf16* __restrict__ A, int lda, long sAb,
    const __bf16* __restrict__ BT, int ldb, long sBb,
    const float* __restrict__ bias,
    void* __restrict__ out0, void* __restrict__ out1, void* __restrict__ out2,
    const float* __restrict__ extra, long sOb,
    const float* __restrict__ scal,
    int Ncols, int K)
{
    constexpr int NSUB   = TN / 32;          // 16-col subtiles per wave
    constexpr unsigned LDS_A0 = 0u;
    constexpr unsigned LDS_A1 = 8192u;
    constexpr unsigned LDS_B0 = 16384u;
    constexpr unsigned LDS_B1 = 16384u + (unsigned)TN * 80u;
    __shared__ __align__(16) char smem[16384 + 2 * TN * 80];

    const int t    = threadIdx.x;
    const int wave = t >> 5;
    const int lane = t & 31;
    const int hs   = lane >> 4;
    const int lr   = lane & 15;
    const int z    = blockIdx.z;
    const long row0 = (long)blockIdx.x * 128;
    const int  c0   = blockIdx.y * TN;
    const int  mi   = (wave & 3) * 32;       // 32-row slice of the 128-row tile
    const int  ni   = (wave >> 2) * (TN / 2);

    const char* Ab = (const char*)(A  + z * sAb);
    const char* Bb = (const char*)(BT + z * sBb);

    // keep the compiler from assuming TDM-written LDS is never stored
    asm volatile("" :: "v"(smem) : "memory");

    v8f acc[2][NSUB];
#pragma unroll
    for (int a = 0; a < 2; ++a)
#pragma unroll
        for (int b = 0; b < NSUB; ++b)
#pragma unroll
            for (int j = 0; j < 8; ++j) acc[a][b][j] = 0.f;

    // prologue: DMA first K-slice into buffer 0
    if (wave == 0)
        tdm_load_2d(Ab + row0 * (long)lda * 2, LDS_A0, 128, (unsigned)(lda >> 1), false);
    if (wave == 1)
        tdm_load_2d(Bb + (long)c0 * ldb * 2, LDS_B0, TN, (unsigned)(ldb >> 1), true);

    for (int k0 = 0, it = 0; k0 < K; k0 += 32, ++it) {
        const int p = it & 1;
        const bool more = (k0 + 32) < K;
        if (wave == 0) {
            if (more) {
                tdm_load_2d(Ab + (row0 * (long)lda + (k0 + 32)) * 2,
                            p ? LDS_A0 : LDS_A1, 128, (unsigned)(lda >> 1), false);
                __builtin_amdgcn_s_wait_tensorcnt(1);
            } else {
                __builtin_amdgcn_s_wait_tensorcnt(0);
            }
        } else if (wave == 1) {
            if (more) {
                tdm_load_2d(Bb + ((long)c0 * ldb + (k0 + 32)) * 2,
                            p ? LDS_B0 : LDS_B1, TN, (unsigned)(ldb >> 1), true);
                __builtin_amdgcn_s_wait_tensorcnt(1);
            } else {
                __builtin_amdgcn_s_wait_tensorcnt(0);
            }
        }
        __syncthreads();

        const char* bufA = smem + (p ? LDS_A1 : LDS_A0);
        const char* bufB = smem + (p ? LDS_B1 : LDS_B0);

        DualU4 f;
        v16bf va[2], vb[NSUB];
#pragma unroll
        for (int sm = 0; sm < 2; ++sm) {
            int r = mi + sm * 16 + lr;
            f.lo = *(const u32x4*)(bufA + r * 64 + hs * 16);
            f.hi = *(const u32x4*)(bufA + r * 64 + 32 + hs * 16);
            va[sm] = __builtin_bit_cast(v16bf, f);
        }
#pragma unroll
        for (int sn = 0; sn < NSUB; ++sn) {
            int c = ni + sn * 16 + lr;
            f.lo = *(const u32x4*)(bufB + c * 80 + hs * 16);
            f.hi = *(const u32x4*)(bufB + c * 80 + 32 + hs * 16);
            vb[sn] = __builtin_bit_cast(v16bf, f);
        }
#pragma unroll
        for (int sm = 0; sm < 2; ++sm)
#pragma unroll
            for (int sn = 0; sn < NSUB; ++sn)
                acc[sm][sn] = __builtin_amdgcn_wmma_f32_16x16x32_bf16(
                    false, va[sm], false, vb[sn], (short)0, acc[sm][sn],
                    false, false);
        __syncthreads();
    }

    // ---- epilogue ----
    const float gb = (EPI == 2) ? scal[8] : 0.f;
#pragma unroll
    for (int sm = 0; sm < 2; ++sm) {
#pragma unroll
        for (int sn = 0; sn < NSUB; ++sn) {
            int col = c0 + ni + sn * 16 + lr;
#pragma unroll
            for (int j = 0; j < 8; ++j) {
                long row = row0 + mi + sm * 16 + hs * 8 + j;
                float v = acc[sm][sn][j];
                if (EPI == 0) {
                    ((__bf16*)out0)[row * Ncols + col] = f2bf(v + bias[col]);
                } else if (EPI == 1) {
                    float val = v + bias[col];
                    int sec = col >> 8, cc = col & 255;
                    if (sec == 0) {                       // qf (f32, row-major)
                        ((float*)out0)[row * HID + cc] = eluplus(val);
                    } else if (sec == 1) {                // kf (bf16, row-major)
                        ((__bf16*)out1)[row * HID + cc] = f2bf(eluplus(val));
                    } else {                              // v  (bf16, TRANSPOSED)
                        long bq = row >> 11, n = row & 2047;
                        ((__bf16*)out2)[((bq << 8) + cc) * (long)NN + n] = f2bf(val);
                    }
                } else if (EPI == 2) {
                    long o = (long)z * sOb + row * Ncols + col;
                    ((__bf16*)out0)[o] = f2bf((1.f - gb) * extra[o] + gb * v);
                } else { // EPI == 3
                    long o = row * Ncols + col;
                    ((float*)out0)[o] = v + bias[col] + extra[o];
                }
            }
        }
    }
}

// ---------------------------------------------------------------------------
// K4: kv[b,h,d,e] = sum_n kf[b,n,h*32+d] * v[b,h*32+e,n] ; ksum[b,h,d]
// (kf row-major [B,N,256]; v transposed [B,256,N])
// ---------------------------------------------------------------------------
__global__ __launch_bounds__(256) void kv_reduce_kernel(
    const __bf16* __restrict__ kfb, const __bf16* __restrict__ vT,
    float* __restrict__ kv, float* __restrict__ ksum)
{
    int bh = blockIdx.x;
    int b = bh >> 3, h = bh & 7;
    const long kbase = (long)b * NN * HID + h * DH;
    const long vbase = ((long)b * HID + h * DH) * NN;
    __shared__ float sk[8][DH];
    __shared__ float sv[8][DH];
    int t = threadIdx.x;
    int li = t >> 5, ld = t & 31;
    int d = t >> 3, e0 = (t & 7) * 4;
    float a0 = 0.f, a1 = 0.f, a2 = 0.f, a3 = 0.f, as = 0.f;
    for (int n0 = 0; n0 < NN; n0 += 8) {
        sk[li][ld] = bf2f(kfb[kbase + (long)(n0 + li) * HID + ld]);
        sv[li][ld] = bf2f(vT [vbase + (long)ld * NN + (n0 + li)]);
        __syncthreads();
#pragma unroll
        for (int i = 0; i < 8; ++i) {
            float kk = sk[i][d];
            a0 += kk * sv[i][e0 + 0];
            a1 += kk * sv[i][e0 + 1];
            a2 += kk * sv[i][e0 + 2];
            a3 += kk * sv[i][e0 + 3];
            if (e0 == 0) as += kk;
        }
        __syncthreads();
    }
    long kb = ((long)bh * DH + d) * DH + e0;
    kv[kb + 0] = a0; kv[kb + 1] = a1; kv[kb + 2] = a2; kv[kb + 3] = a3;
    if (e0 == 0) ksum[bh * DH + d] = as + 1e-8f;
}

// ---------------------------------------------------------------------------
// K5: linear attention normalize: lin[row,col] = (qf . kv[:,e]) / (qf . ksum)
// ---------------------------------------------------------------------------
__global__ __launch_bounds__(256) void lin_attn_kernel(
    const float* __restrict__ qf, const float* __restrict__ kv,
    const float* __restrict__ ksum, float* __restrict__ lin)
{
    long row = blockIdx.x;                 // b*N + n
    int b = (int)(row / NN);
    int t = threadIdx.x;
    __shared__ float sq[HID];
    __shared__ float sden[HEADS];
    sq[t] = qf[row * HID + t];
    __syncthreads();
    if (t < HEADS) {
        const float* ks = ksum + (b * HEADS + t) * DH;
        float den = 1e-8f;
#pragma unroll
        for (int d = 0; d < DH; ++d) den += sq[t * DH + d] * ks[d];
        sden[t] = den;
    }
    __syncthreads();
    int h = t >> 5, e = t & 31;
    const float* kvp = kv + ((long)(b * HEADS + h) * DH) * DH + e;
    float num = 0.f;
#pragma unroll
    for (int d = 0; d < DH; ++d) num += sq[h * DH + d] * kvp[d * DH];
    lin[row * HID + t] = num / sden[h];
}

// ---------------------------------------------------------------------------
// K9: LayerNorm over HID, write final f32 output
// ---------------------------------------------------------------------------
__global__ __launch_bounds__(256) void ln_kernel(
    const float* __restrict__ y, const float* __restrict__ gamma,
    const float* __restrict__ beta, float* __restrict__ out)
{
    long row = blockIdx.x;
    int t = threadIdx.x;
    __shared__ float red[HID];
    float v = y[row * HID + t];
    red[t] = v; __syncthreads();
    for (int s = 128; s > 0; s >>= 1) { if (t < s) red[t] += red[t + s]; __syncthreads(); }
    float mu = red[0] * (1.f / HID);
    __syncthreads();
    float dv = v - mu;
    red[t] = dv * dv; __syncthreads();
    for (int s = 128; s > 0; s >>= 1) { if (t < s) red[t] += red[t + s]; __syncthreads(); }
    float var = red[0] * (1.f / HID);
    out[row * HID + t] = dv * rsqrtf(var + 1e-5f) * gamma[t] + beta[t];
}

// ---------------------------------------------------------------------------
// Launcher
// ---------------------------------------------------------------------------
extern "C" void kernel_launch(void* const* d_in, const int* in_sizes, int n_in,
                              void* d_out, int out_size, void* d_ws, size_t ws_size,
                              hipStream_t stream) {
    const float* x          = (const float*)d_in[0];
    const float* features   = (const float*)d_in[1];
    const float* delay_log  = (const float*)d_in[2];
    const float* w_dp       = (const float*)d_in[3];
    const float* b_dp       = (const float*)d_in[4];
    const float* w_qkv_low  = (const float*)d_in[5];
    const float* b_qkv_low  = (const float*)d_in[6];
    const float* w_qkv_high = (const float*)d_in[7];
    const float* b_qkv_high = (const float*)d_in[8];
    const float* w_out_low  = (const float*)d_in[9];
    const float* b_out_low  = (const float*)d_in[10];
    const float* w_out_high = (const float*)d_in[11];
    const float* b_out_high = (const float*)d_in[12];
    const float* adj        = (const float*)d_in[13];
    const float* geo_w      = (const float*)d_in[14];
    const float* ln_gamma   = (const float*)d_in[15];
    const float* ln_beta    = (const float*)d_in[16];
    float* outp = (float*)d_out;

    char* w = (char*)d_ws;
    auto alloc = [&](size_t bytes) -> char* {
        char* p = w; w += (bytes + 255) & ~(size_t)255; return p;
    };
    float*  scal   = (float*)alloc(256);
    __bf16* wqlT   = (__bf16*)alloc((size_t)S1 * 2);
    __bf16* wqhT   = (__bf16*)alloc((size_t)S2 * 2);
    __bf16* wolT   = (__bf16*)alloc((size_t)S3 * 2);
    __bf16* wohT   = (__bf16*)alloc((size_t)S4 * 2);
    __bf16* adjb   = (__bf16*)alloc((size_t)S5 * 2);
    __bf16* comb   = (__bf16*)alloc((size_t)ROWS * HID * 2);
    __bf16* qkvmid = (__bf16*)alloc((size_t)ROWS * 3 * BOT * 2);
    float*  qf     = (float*)alloc((size_t)ROWS * HID * 4);
    __bf16* kfb    = (__bf16*)alloc((size_t)ROWS * HID * 2);
    __bf16* vT     = (__bf16*)alloc((size_t)ROWS * HID * 2);   // [B,256,2048]
    float*  kv     = (float*)alloc((size_t)BB * HEADS * DH * DH * 4);
    float*  ksum   = (float*)alloc((size_t)BB * HEADS * DH * 4);
    float*  lin    = (float*)alloc((size_t)ROWS * HID * 4);    // reused as y
    __bf16* attb   = (__bf16*)alloc((size_t)ROWS * HID * 2);
    __bf16* outmid = (__bf16*)alloc((size_t)ROWS * BOT * 2);

    // K0: weight conversion (transposed) + scalars
    prep_kernel<<<(STOT + 255) / 256, 256, 0, stream>>>(
        delay_log, geo_w, w_qkv_low, w_qkv_high, w_out_low, w_out_high, adj,
        scal, wqlT, wqhT, wolT, wohT, adjb);

    // K1: delay-fused combined
    combined_kernel<<<ROWS, 256, 0, stream>>>(x, features, w_dp, b_dp, scal, comb);

    // K2: qkv_mid = combined @ w_qkv_low + b  [65536,192] (bf16), TN=64
    gemm_bf16_kernel<0, 64><<<dim3(ROWS / 128, 3, 1), 256, 0, stream>>>(
        comb, HID, 0, wqlT, HID, 0, b_qkv_low,
        qkvmid, nullptr, nullptr, nullptr, 0, scal, 3 * BOT, HID);

    // K3: qkv = qkv_mid @ w_qkv_high + b, split into qf/kf (elu+1) and v^T, TN=128
    gemm_bf16_kernel<1, 128><<<dim3(ROWS / 128, 6, 1), 256, 0, stream>>>(
        qkvmid, 3 * BOT, 0, wqhT, 3 * BOT, 0, b_qkv_high,
        qf, kfb, vT, nullptr, 0, scal, 3 * HID, 3 * BOT);

    // K4: kv / k_sum reductions
    kv_reduce_kernel<<<BB * HEADS, 256, 0, stream>>>(kfb, vT, kv, ksum);

    // K5: linear attention output
    lin_attn_kernel<<<ROWS, 256, 0, stream>>>(qf, kv, ksum, lin);

    // K6: geo GEMM (batched adj @ V_b via V^T) + blend -> attended (bf16), TN=128
    gemm_bf16_kernel<2, 128><<<dim3(NN / 128, HID / 128, BB), 256, 0, stream>>>(
        adjb, NN, 0, vT, NN, (long)HID * NN, nullptr,
        attb, nullptr, nullptr, lin, (long)NN * HID, scal, HID, NN);

    // K7: out_mid = attended @ w_out_low + b  [65536,64] (bf16), TN=64
    gemm_bf16_kernel<0, 64><<<dim3(ROWS / 128, 1, 1), 256, 0, stream>>>(
        attb, HID, 0, wolT, HID, 0, b_out_low,
        outmid, nullptr, nullptr, nullptr, 0, scal, BOT, HID);

    // K8: y = out_mid @ w_out_high + b + features  (f32, reuse lin), TN=128
    gemm_bf16_kernel<3, 128><<<dim3(ROWS / 128, HID / 128, 1), 256, 0, stream>>>(
        outmid, BOT, 0, wohT, BOT, 0, b_out_high,
        lin, nullptr, nullptr, features, 0, scal, HID, BOT);

    // K9: LayerNorm -> d_out
    ln_kernel<<<ROWS, 256, 0, stream>>>(lin, ln_gamma, ln_beta, outp);
}
